// HCNetFull_16131897164407
// MI455X (gfx1250) — matrix operand order repaired and checked by reading, not compiled, not run on hardware
//
#include <hip/hip_runtime.h>
#include <hip/hip_bf16.h>
#include <math.h>

// ---------------------------------------------------------------------------
// MI455X (gfx1250) fused HCNet forward, v2.
// 32-token tiles: each B fragment (global_load_b128 pair from L2) feeds TWO
// v_wmma_f32_16x16x32_f16 ops (M-tiles 0..15 and 16..31), halving the L2
// bytes/FLOP of the dominant fc1/fc2 GEMMs vs v1.
// ---------------------------------------------------------------------------

typedef __attribute__((ext_vector_type(16))) _Float16 v16h;
typedef __attribute__((ext_vector_type(8)))  _Float16 v8h;
typedef __attribute__((ext_vector_type(8)))  float    v8f;

#define TM   32      // tokens per workgroup tile
#define DH   512     // hidden dim
#define DF   1024    // ffn dim
#define NLAY 8

__device__ __forceinline__ float gelu_exact(float v) {
    return 0.5f * v * (1.0f + erff(v * 0.70710678118654752f));
}

// reduce across the 16 lanes that share one token (two tokens per wave32)
__device__ __forceinline__ float sum16(float v) {
    v += __shfl_xor(v, 1);
    v += __shfl_xor(v, 2);
    v += __shfl_xor(v, 4);
    v += __shfl_xor(v, 8);
    return v;
}

// LayerNorm over 512 features for a 16-token half-tile (16 threads/token).
__device__ __forceinline__ void ln_to_f16(const float* __restrict__ in,
                                          _Float16* __restrict__ out,
                                          const float* __restrict__ g,
                                          const float* __restrict__ b) {
    const int t = threadIdx.x >> 4, s = threadIdx.x & 15;
    float v[32];
    float sum = 0.f;
#pragma unroll
    for (int i = 0; i < 32; ++i) { v[i] = in[t * DH + i * 16 + s]; sum += v[i]; }
    const float mean = sum16(sum) * (1.0f / DH);
    float var = 0.f;
#pragma unroll
    for (int i = 0; i < 32; ++i) { float d = v[i] - mean; var += d * d; }
    const float rstd = rsqrtf(sum16(var) * (1.0f / DH) + 1e-5f);
#pragma unroll
    for (int i = 0; i < 32; ++i) {
        const int c = i * 16 + s;
        out[t * DH + c] = (_Float16)((v[i] - mean) * rstd * g[c] + b[c]);
    }
}

__device__ __forceinline__ void ln_inplace(float* __restrict__ buf,
                                           const float* __restrict__ g,
                                           const float* __restrict__ b) {
    const int t = threadIdx.x >> 4, s = threadIdx.x & 15;
    float v[32];
    float sum = 0.f;
#pragma unroll
    for (int i = 0; i < 32; ++i) { v[i] = buf[t * DH + i * 16 + s]; sum += v[i]; }
    const float mean = sum16(sum) * (1.0f / DH);
    float var = 0.f;
#pragma unroll
    for (int i = 0; i < 32; ++i) { float d = v[i] - mean; var += d * d; }
    const float rstd = rsqrtf(sum16(var) * (1.0f / DH) + 1e-5f);
#pragma unroll
    for (int i = 0; i < 32; ++i) {
        const int c = i * 16 + s;
        buf[t * DH + c] = (v[i] - mean) * rstd * g[c] + b[c];
    }
}

// Dual-M 16x16 output tiles of [32 x K] (LDS f16 row major) @ BT[N x K] (global f16).
// One B fragment (two 8-half chunks per lane, same k order as A) feeds two WMMAs.
// A fragment layout (16-bit A 16x32): lane m=lane&15; halves 0..7 -> K base+koff+0..7,
// halves 8..15 -> K base+koff+16..23, koff = 8*(lane>>4)  => two contiguous 8-half chunks.
__device__ __forceinline__ void wmma_rowtile2(const _Float16* __restrict__ ldsA,
                                              const _Float16* __restrict__ BT,
                                              int K, int nt, int lane,
                                              v8f& acc0, v8f& acc1) {
    const int mlo = lane & 15;
    const int hi  = lane >> 4;
    const _Float16* arow0 = ldsA + mlo * K + hi * 8;          // tokens 0..15
    const _Float16* arow1 = arow0 + 16 * K;                   // tokens 16..31
    const _Float16* brow  = BT + (size_t)(nt * 16 + mlo) * K + hi * 8;
    acc0 = (v8f){};
    acc1 = (v8f){};
    for (int ks = 0; ks < K; ks += 32) {
        union { v16h v; v8h h[2]; } a0, a1, b;
        b.h[0]  = *(const v8h*)(brow + ks);
        b.h[1]  = *(const v8h*)(brow + ks + 16);
        a0.h[0] = *(const v8h*)(arow0 + ks);
        a0.h[1] = *(const v8h*)(arow0 + ks + 16);
        a1.h[0] = *(const v8h*)(arow1 + ks);
        a1.h[1] = *(const v8h*)(arow1 + ks + 16);
        acc0 = __builtin_amdgcn_wmma_f32_16x16x32_f16(false, a0.v, false, b.v,
                                                      (short)0, acc0, false, false);
        acc1 = __builtin_amdgcn_wmma_f32_16x16x32_f16(false, a1.v, false, b.v,
                                                      (short)0, acc1, false, false);
    }
}

// --------------------------- weight f32 -> f16 transpose --------------------
__global__ void convT_kernel(const float* __restrict__ in, _Float16* __restrict__ out,
                             int rows, int cols) {
    const int n = rows * cols;
    for (int idx = blockIdx.x * blockDim.x + threadIdx.x; idx < n;
         idx += gridDim.x * blockDim.x) {
        const int r = idx / cols, c = idx % cols;
        out[(size_t)c * rows + r] = (_Float16)in[idx];
    }
}

// ------------------------------- fused forward ------------------------------
__global__ __launch_bounds__(256) void fused_forward(
    const float* __restrict__ x,
    const float* __restrict__ in_w,  const float* __restrict__ in_b,
    const float* __restrict__ fc1_b, const float* __restrict__ fc2_b,
    const float* __restrict__ geo_w, const float* __restrict__ geo_b,
    const float* __restrict__ n1_g,  const float* __restrict__ n1_b,
    const float* __restrict__ n2_g,  const float* __restrict__ n2_b,
    const float* __restrict__ gi_pos_w, const float* __restrict__ gi_pos_b,
    const float* __restrict__ gi_vel_w, const float* __restrict__ gi_vel_b,
    const float* __restrict__ gi_int_b,
    const float* __restrict__ gi_n_g, const float* __restrict__ gi_n_b,
    const float* __restrict__ pi1_b,  const float* __restrict__ pi2_b,
    const float* __restrict__ out_w,  const float* __restrict__ out_b,
    const _Float16* __restrict__ fc1T, const _Float16* __restrict__ fc2T,
    const _Float16* __restrict__ giT,  const _Float16* __restrict__ pi1T,
    const _Float16* __restrict__ pi2T,
    float* __restrict__ out) {

    __shared__ float    hbuf[TM * DH];                        // residual h  (64 KB)
    __shared__ _Float16 stageA[TM * DH];                      // A operand   (32 KB)
    __shared__ __align__(32) unsigned char ubuf[TM * DF * 2]; // stageB f16 / ybuf f32 (64 KB)
    __shared__ float    gwbuf[512];                           // geo weight slice (2 KB)
    _Float16* stageB = (_Float16*)ubuf;
    float*    ybuf   = (float*)ubuf;

    const int wave = threadIdx.x >> 5;
    const int lane = threadIdx.x & 31;
    const int mlo  = lane & 15;
    const int hi   = lane >> 4;
    const int tok0 = blockIdx.x * TM;

    // ---- input projection: h = x @ in_w + in_b  ([32x4] @ [4x512]) ----
#pragma unroll
    for (int tt = 0; tt < 2; ++tt) {
        const int t = (threadIdx.x >> 4) + tt * 16, s = threadIdx.x & 15;
        float xv[4];
#pragma unroll
        for (int k = 0; k < 4; ++k) xv[k] = x[(size_t)(tok0 + t) * 4 + k];
#pragma unroll
        for (int i = 0; i < 32; ++i) {
            const int c = i * 16 + s;
            float a = in_b[c];
#pragma unroll
            for (int k = 0; k < 4; ++k) a += xv[k] * in_w[k * DH + c];
            hbuf[t * DH + c] = a;
        }
    }
    __syncthreads();

    // ------------------------------ layer stack -----------------------------
    for (int l = 0; l < NLAY; ++l) {
        // cache this layer's geo weights in LDS; LN1 (both token halves) -> f16
        for (int i = threadIdx.x; i < 512; i += 256) gwbuf[i] = geo_w[l * 512 + i];
        ln_to_f16(hbuf, stageA, n1_g + l * DH, n1_b + l * DH);
        ln_to_f16(hbuf + 16 * DH, stageA + 16 * DH, n1_g + l * DH, n1_b + l * DH);
        if (l + 1 < NLAY && lane == 0) {   // warm next layer's panels in L2
            __builtin_prefetch(fc1T + (size_t)(l + 1) * DF * DH, 0, 1);
            __builtin_prefetch(fc2T + (size_t)(l + 1) * DH * DF, 0, 1);
        }
        __syncthreads();

        // GEMM1: [32x512] @ [512x1024], +bias, exact GELU -> stageB f16
        const _Float16* W1 = fc1T + (size_t)l * DF * DH;   // [1024][512]
        for (int nt = wave; nt < DF / 16; nt += 8) {
            v8f acc0, acc1;
            wmma_rowtile2(stageA, W1, DH, nt, lane, acc0, acc1);
            const int n = nt * 16 + mlo;
            const float bias = fc1_b[l * DF + n];
#pragma unroll
            for (int r = 0; r < 8; ++r) {
                stageB[(r + hi * 8) * DF + n]      = (_Float16)gelu_exact(acc0[r] + bias);
                stageB[(r + hi * 8 + 16) * DF + n] = (_Float16)gelu_exact(acc1[r] + bias);
            }
        }
        __syncthreads();

        // GEMM2: [32x1024] @ [1024x512]; hold accs in VGPRs across the barrier,
        // then overwrite the stageB region with f32 y = fc2(out)+bias+h.
        const _Float16* W2 = fc2T + (size_t)l * DH * DF;   // [512][1024]
        v8f acc2[4][2];
#pragma unroll
        for (int i = 0; i < 4; ++i)
            wmma_rowtile2(stageB, W2, DF, wave * 4 + i, lane, acc2[i][0], acc2[i][1]);
        __syncthreads();
#pragma unroll
        for (int i = 0; i < 4; ++i) {
            const int nt = wave * 4 + i;
            const int n  = nt * 16 + mlo;
            const float bias = fc2_b[l * DH + n];
#pragma unroll
            for (int r = 0; r < 8; ++r) {
                const int m0 = r + hi * 8;
                const int m1 = m0 + 16;
                ybuf[m0 * DH + n] = acc2[i][0][r] + bias + hbuf[m0 * DH + n];
                ybuf[m1 * DH + n] = acc2[i][1][r] + bias + hbuf[m1 * DH + n];
            }
        }
        __syncthreads();

        // geometric group mixing: geo[k] = sum_ij xg[i]*xg[j]*gw[i][j][k] + gb
        for (int task = threadIdx.x; task < TM * 64; task += 256) {
            const int t = task >> 6, g = task & 63;
            float xg[8], geo[8];
#pragma unroll
            for (int i = 0; i < 8; ++i) { xg[i] = ybuf[t * DH + g * 8 + i]; geo[i] = 0.f; }
            for (int i = 0; i < 8; ++i) {
#pragma unroll
                for (int j = 0; j < 8; ++j) {
                    const float p = xg[i] * xg[j];
                    const float* gw = gwbuf + (i * 8 + j) * 8;
#pragma unroll
                    for (int k = 0; k < 8; ++k) geo[k] += p * gw[k];
                }
            }
#pragma unroll
            for (int k = 0; k < 8; ++k)
                hbuf[t * DH + g * 8 + k] =
                    ybuf[t * DH + g * 8 + k] + 0.1f * (geo[k] + geo_b[l * 8 + k]);
        }
        __syncthreads();
        ln_inplace(hbuf, n2_g + l * DH, n2_b + l * DH);
        ln_inplace(hbuf + 16 * DH, n2_g + l * DH, n2_b + l * DH);
        __syncthreads();
    }

    // --------------------- GeometricInteraction head ------------------------
    // pos/vel: 32 tok x 8 outputs x 2 -> 512 dot-512 products, 2 per thread
    for (int p = threadIdx.x; p < 512; p += 256) {
        const int which = p >> 8;                    // 0 = pos, 1 = vel
        const int rem   = p & 255;
        const int t = rem >> 3, o = rem & 7;
        const float* W = which ? gi_vel_w : gi_pos_w;
        float a = which ? gi_vel_b[o] : gi_pos_b[o];
        for (int k = 0; k < DH; ++k) a += hbuf[t * DH + k] * W[k * 8 + o];
        ybuf[which * 256 + rem] = a;                 // pos[32][8] | vel[32][8]
    }
    __syncthreads();
    // inter = outer(pos, vel) -> f16 A tile [32 x 64]
    for (int task = threadIdx.x; task < TM * 64; task += 256) {
        const int t = task >> 6, g = task & 63, i = g >> 3, j = g & 7;
        stageA[t * 64 + g] = (_Float16)(ybuf[t * 8 + i] * ybuf[256 + t * 8 + j]);
    }
    __syncthreads();
    // inter @ gi_int_w : K=64, N=512, accumulate into h
#pragma unroll
    for (int i = 0; i < 4; ++i) {
        const int nt = wave * 4 + i;
        v8f acc0, acc1;
        wmma_rowtile2(stageA, giT, 64, nt, lane, acc0, acc1);
        const int n = nt * 16 + mlo;
#pragma unroll
        for (int r = 0; r < 8; ++r) {
            const int m0 = r + hi * 8;
            const int m1 = m0 + 16;
            hbuf[m0 * DH + n] += acc0[r] + gi_int_b[n];
            hbuf[m1 * DH + n] += acc1[r] + gi_int_b[n];
        }
    }
    __syncthreads();
    ln_inplace(hbuf, gi_n_g, gi_n_b);
    ln_inplace(hbuf + 16 * DH, gi_n_g, gi_n_b);
    __syncthreads();

    // ----------------------- particle interaction MLP -----------------------
    for (int i = threadIdx.x; i < TM * DH; i += 256) stageA[i] = (_Float16)hbuf[i];
    __syncthreads();
#pragma unroll
    for (int i = 0; i < 4; ++i) {
        const int nt = wave * 4 + i;
        v8f acc0, acc1;
        wmma_rowtile2(stageA, pi1T, DH, nt, lane, acc0, acc1);
        const int n = nt * 16 + mlo;
        const float bias = pi1_b[n];
#pragma unroll
        for (int r = 0; r < 8; ++r) {
            stageB[(r + hi * 8) * DH + n]      = (_Float16)gelu_exact(acc0[r] + bias);
            stageB[(r + hi * 8 + 16) * DH + n] = (_Float16)gelu_exact(acc1[r] + bias);
        }
    }
    __syncthreads();
    v8f accp[4][2];
#pragma unroll
    for (int i = 0; i < 4; ++i)
        wmma_rowtile2(stageB, pi2T, DH, wave * 4 + i, lane, accp[i][0], accp[i][1]);
    __syncthreads();
#pragma unroll
    for (int i = 0; i < 4; ++i) {
        const int nt = wave * 4 + i;
        const int n  = nt * 16 + mlo;
#pragma unroll
        for (int r = 0; r < 8; ++r) {
            const int m0 = r + hi * 8;
            const int m1 = m0 + 16;
            ybuf[m0 * DH + n] = accp[i][0][r] + pi2_b[n];
            ybuf[m1 * DH + n] = accp[i][1][r] + pi2_b[n];
        }
    }
    __syncthreads();

    // ------------------------ output head + residual ------------------------
    if (threadIdx.x < 128) {
        const int t = threadIdx.x >> 2, c = threadIdx.x & 3;
        float a = out_b[c];
        for (int k = 0; k < DH; ++k) a += ybuf[t * DH + k] * out_w[k * 4 + c];
        out[(size_t)(tok0 + t) * 4 + c] = x[(size_t)(tok0 + t) * 4 + c] + a;
    }
}

// ------------------------------- host launch --------------------------------
extern "C" void kernel_launch(void* const* d_in, const int* in_sizes, int n_in,
                              void* d_out, int out_size, void* d_ws, size_t ws_size,
                              hipStream_t stream) {
    const float* x        = (const float*)d_in[0];
    const float* in_w     = (const float*)d_in[1];
    const float* in_b     = (const float*)d_in[2];
    const float* fc1_w    = (const float*)d_in[3];
    const float* fc1_b    = (const float*)d_in[4];
    const float* fc2_w    = (const float*)d_in[5];
    const float* fc2_b    = (const float*)d_in[6];
    const float* geo_w    = (const float*)d_in[7];
    const float* geo_b    = (const float*)d_in[8];
    const float* n1_g     = (const float*)d_in[9];
    const float* n1_b     = (const float*)d_in[10];
    const float* n2_g     = (const float*)d_in[11];
    const float* n2_b     = (const float*)d_in[12];
    const float* gi_pos_w = (const float*)d_in[13];
    const float* gi_pos_b = (const float*)d_in[14];
    const float* gi_vel_w = (const float*)d_in[15];
    const float* gi_vel_b = (const float*)d_in[16];
    const float* gi_int_w = (const float*)d_in[17];
    const float* gi_int_b = (const float*)d_in[18];
    const float* gi_n_g   = (const float*)d_in[19];
    const float* gi_n_b   = (const float*)d_in[20];
    const float* pi1_w    = (const float*)d_in[21];
    const float* pi1_b    = (const float*)d_in[22];
    const float* pi2_w    = (const float*)d_in[23];
    const float* pi2_b    = (const float*)d_in[24];
    const float* out_w    = (const float*)d_in[25];
    const float* out_b    = (const float*)d_in[26];
    float* out = (float*)d_out;

    // workspace layout: f16 transposed weights (BT = [N][K])
    _Float16* fc1T = (_Float16*)d_ws;                           // 8 x 1024 x 512
    _Float16* fc2T = fc1T + (size_t)NLAY * DF * DH;             // 8 x 512 x 1024
    _Float16* giT  = fc2T + (size_t)NLAY * DH * DF;             // 512 x 64
    _Float16* pi1T = giT  + (size_t)DH * 64;                    // 512 x 512
    _Float16* pi2T = pi1T + (size_t)DH * DH;                    // 512 x 512

    for (int l = 0; l < NLAY; ++l) {
        convT_kernel<<<512, 256, 0, stream>>>(fc1_w + (size_t)l * DH * DF,
                                              fc1T + (size_t)l * DF * DH, DH, DF);
        convT_kernel<<<512, 256, 0, stream>>>(fc2_w + (size_t)l * DF * DH,
                                              fc2T + (size_t)l * DH * DF, DF, DH);
    }
    convT_kernel<<<64,  256, 0, stream>>>(gi_int_w, giT, 64, DH);
    convT_kernel<<<256, 256, 0, stream>>>(pi1_w, pi1T, DH, DH);
    convT_kernel<<<256, 256, 0, stream>>>(pi2_w, pi2T, DH, DH);

    const int tokens = 32 * 1024;            // B * N
    fused_forward<<<tokens / TM, 256, 0, stream>>>(
        x, in_w, in_b, fc1_b, fc2_b, geo_w, geo_b,
        n1_g, n1_b, n2_g, n2_b,
        gi_pos_w, gi_pos_b, gi_vel_w, gi_vel_b,
        gi_int_b, gi_n_g, gi_n_b,
        pi1_b, pi2_b, out_w, out_b,
        fc1T, fc2T, giT, pi1T, pi2T, out);
}
